// DecoderBlock_9251359556061
// MI455X (gfx1250) — compile-verified, matrix operand order checked by reading, not run on hardware
//
#include <hip/hip_runtime.h>

// ---------------- problem constants ----------------
#define H_DIM 1024
#define NH    16
#define DKD   64
#define BATCH 4
#define SEQ   1024
#define MROWS (BATCH * SEQ)   // 4096

typedef _Float16 v8h  __attribute__((ext_vector_type(8)));
typedef _Float16 v16h __attribute__((ext_vector_type(16)));
typedef float    v8f  __attribute__((ext_vector_type(8)));
typedef int      v4i  __attribute__((ext_vector_type(4)));

// ---- optional CDNA5 async global->LDS path (ASYNCcnt), feature-gated ------
#if defined(__has_builtin)
#  if __has_builtin(__builtin_amdgcn_global_load_async_to_lds_b128) && \
      __has_builtin(__builtin_amdgcn_s_wait_asynccnt)
#    define USE_ASYNC_LDS 1
#  endif
#endif
#ifndef USE_ASYNC_LDS
#  define USE_ASYNC_LDS 0
#endif

#if USE_ASYNC_LDS
typedef __attribute__((address_space(1))) v4i* as1_v4i;
typedef __attribute__((address_space(3))) v4i* as3_v4i;
#endif

// one 16B chunk per lane: global -> LDS
__device__ __forceinline__ void cp_b128(const _Float16* g, _Float16* l) {
#if USE_ASYNC_LDS
  __builtin_amdgcn_global_load_async_to_lds_b128((as1_v4i)g, (as3_v4i)l, 0, 0);
#else
  *(v8h*)l = *(const v8h*)g;
#endif
}
__device__ __forceinline__ void cp_wait() {
#if USE_ASYNC_LDS
  __builtin_amdgcn_s_wait_asynccnt(0);
#endif
}

// A/B fragment (16x32 f16, wave32): lane holds row lane%16; K elements are
// {koff..koff+7, koff+16..koff+23}, koff = (lane<16)?0:8  ->  two b128 LDS loads.
__device__ __forceinline__ v16h ld_frag(const _Float16* p) {
  v8h lo = *(const v8h*)(p);
  v8h hi = *(const v8h*)(p + 16);
  return __builtin_shufflevector(lo, hi, 0, 1, 2, 3, 4, 5, 6, 7,
                                 8, 9, 10, 11, 12, 13, 14, 15);
}

__device__ __forceinline__ v8f wmma_f16(v16h a, v16h b, v8f c) {
  // D = A x B + C, 16x16x32, f32 accumulate
  return __builtin_amdgcn_wmma_f32_16x16x32_f16(false, a, false, b, (short)0, c,
                                                false, false);
}

// ---------------- weight transpose+convert: wt[n*K+k] = (f16)w[k*N+n] -------
__global__ __launch_bounds__(256) void wcvt_kernel(const float* __restrict__ w,
                                                   _Float16* __restrict__ wt,
                                                   int K, int N) {
  long idx = (long)blockIdx.x * 256 + threadIdx.x;
  if (idx >= (long)K * N) return;
  int n = (int)(idx / K);
  int k = (int)(idx % K);
  wt[idx] = (_Float16)w[(size_t)k * N + n];
}

// ---------------- LayerNorm (f32 in -> f16 out), one row per block ----------
__global__ __launch_bounds__(256) void layernorm_f16_kernel(
    const float* __restrict__ in, const float* __restrict__ g,
    const float* __restrict__ beta, _Float16* __restrict__ out) {
  __shared__ float wsum[8];
  __shared__ float wsq[8];
  const int row = blockIdx.x;
  const int tid = threadIdx.x;
  const float* xr = in + (size_t)row * H_DIM;
  float s = 0.f, ss = 0.f;
  for (int i = tid; i < H_DIM; i += 256) {
    float v = xr[i];
    s += v;
    ss += v * v;
  }
#pragma unroll
  for (int o = 16; o > 0; o >>= 1) {
    s += __shfl_xor(s, o);
    ss += __shfl_xor(ss, o);
  }
  if ((tid & 31) == 0) {
    wsum[tid >> 5] = s;
    wsq[tid >> 5] = ss;
  }
  __syncthreads();
  float ts = 0.f, tss = 0.f;
#pragma unroll
  for (int w = 0; w < 8; ++w) {
    ts += wsum[w];
    tss += wsq[w];
  }
  const float mean = ts * (1.0f / H_DIM);
  const float var = tss * (1.0f / H_DIM) - mean * mean;
  const float inv = rsqrtf(var + 1e-5f);
  for (int i = tid; i < H_DIM; i += 256)
    out[(size_t)row * H_DIM + i] = (_Float16)((xr[i] - mean) * inv * g[i] + beta[i]);
}

// ---------------- WMMA GEMM: out[M,N] = A[M,K] * Bt[N,K]^T (+bias,+res) -----
// Workgroup tile 128x128, 8 waves, each wave 32x64 (2x4 WMMA 16x16 tiles).
#define TM 128
#define TN 128
#define TK 32
#define LK 40  // LDS row stride (halves): 32 data + 8 pad; 80B keeps 16B align

template <bool OUT_F32, bool RELU>
__global__ __launch_bounds__(256, 2) void gemm_wmma_kernel(
    const _Float16* __restrict__ A, const _Float16* __restrict__ Bt,
    const float* __restrict__ bias, const float* __restrict__ res,
    void* __restrict__ outp, int M, int N, int K) {
  __shared__ __align__(16) _Float16 As[TM * LK];
  __shared__ __align__(16) _Float16 Bs[TN * LK];
  const int tid = threadIdx.x;
  const int lane = tid & 31;
  const int wid = tid >> 5;
  const int bm = blockIdx.y * TM;
  const int bn = blockIdx.x * TN;
  const int wm = (wid >> 1) * 32;  // 0,32,64,96
  const int wn = (wid & 1) * 64;   // 0,64
  const int r16 = lane & 15;
  const int koff = (lane & 16) ? 8 : 0;
  const int rhi = (lane & 16) ? 8 : 0;

  const v8f vzero = {};
  v8f acc[2][4];
#pragma unroll
  for (int i = 0; i < 2; ++i)
#pragma unroll
    for (int j = 0; j < 4; ++j) acc[i][j] = vzero;

  const int ldrow = tid >> 2;       // 0..63 (x2 iterations -> 128 rows)
  const int ldseg = (tid & 3) * 8;  // 0,8,16,24

  for (int k0 = 0; k0 < K; k0 += TK) {
#pragma unroll
    for (int i = 0; i < 2; ++i) {
      int row = ldrow + i * 64;
      cp_b128(&A[(size_t)(bm + row) * K + k0 + ldseg], &As[row * LK + ldseg]);
      cp_b128(&Bt[(size_t)(bn + row) * K + k0 + ldseg], &Bs[row * LK + ldseg]);
    }
    if (k0 + TK < K) {  // hint next K-tile into cache (global_prefetch_b8)
      __builtin_prefetch(&A[(size_t)(bm + ldrow) * K + k0 + TK + ldseg], 0, 1);
      __builtin_prefetch(&Bt[(size_t)(bn + ldrow) * K + k0 + TK + ldseg], 0, 1);
    }
    cp_wait();
    __syncthreads();

    v16h af[2], bf[4];
#pragma unroll
    for (int i = 0; i < 2; ++i)
      af[i] = ld_frag(&As[(wm + i * 16 + r16) * LK + koff]);
#pragma unroll
    for (int j = 0; j < 4; ++j)
      bf[j] = ld_frag(&Bs[(wn + j * 16 + r16) * LK + koff]);
#pragma unroll
    for (int i = 0; i < 2; ++i)
#pragma unroll
      for (int j = 0; j < 4; ++j) acc[i][j] = wmma_f16(af[i], bf[j], acc[i][j]);
    __syncthreads();
  }

  // C/D layout: element v of lane -> M = v + 8*(lane>=16), N = lane%16
#pragma unroll
  for (int i = 0; i < 2; ++i)
#pragma unroll
    for (int j = 0; j < 4; ++j)
#pragma unroll
      for (int v = 0; v < 8; ++v) {
        int gm = bm + wm + i * 16 + v + rhi;
        int gn = bn + wn + j * 16 + r16;
        float val = acc[i][j][v];
        if (bias) val += bias[gn];
        if (RELU) val = fmaxf(val, 0.0f);
        size_t idx = (size_t)gm * N + gn;
        if (OUT_F32) {
          float rv = res ? res[idx] : 0.0f;
          ((float*)outp)[idx] = val + rv;
        } else {
          ((_Float16*)outp)[idx] = (_Float16)val;
        }
      }
}

// ---------------- fused attention (flash style), 1 wave per 16-query tile ---
// Q,K,V,O are f16 [B*S, H]; head h occupies columns h*64..h*64+63.
template <bool CAUSAL>
__global__ __launch_bounds__(32) void attention_kernel(
    const _Float16* __restrict__ Qm, const _Float16* __restrict__ Km,
    const _Float16* __restrict__ Vm, _Float16* __restrict__ Om, int S_k) {
  __shared__ __align__(16) _Float16 Qs[16 * 72];  // 16 q-rows x 64 d (+pad)
  __shared__ __align__(16) _Float16 Ks[32 * 72];  // 32 keys x 64 d   (B^T for scores)
  __shared__ __align__(16) _Float16 Vs[64 * 40];  // 64 d x 32 keys   (B^T for P*V)
  __shared__ __align__(16) _Float16 Ps[16 * 40];  // 16 q x 32 keys   (A for P*V)
  const int lane = threadIdx.x;
  const int qbase = blockIdx.x * 16;
  const int h = blockIdx.y;
  const int b = blockIdx.z;
  const int r16 = lane & 15;
  const int koff = (lane & 16) ? 8 : 0;
  const int rhi = (lane & 16) ? 8 : 0;
  const size_t qrow0 = ((size_t)b * SEQ + qbase) * H_DIM + h * DKD;
  const size_t krow0 = ((size_t)b * S_k) * H_DIM + h * DKD;

#pragma unroll
  for (int i = 0; i < 4; ++i) {  // 16x64 halves = 128 v8h chunks
    int c = lane + i * 32;
    int row = c >> 3, seg = (c & 7) * 8;
    cp_b128(&Qm[qrow0 + (size_t)row * H_DIM + seg], &Qs[row * 72 + seg]);
  }

  const v8f vzero = {};
  v8f ctx[4];
#pragma unroll
  for (int t = 0; t < 4; ++t) ctx[t] = vzero;
  float mrow[8], lrow[8];
#pragma unroll
  for (int v = 0; v < 8; ++v) {
    mrow[v] = -1e30f;
    lrow[v] = 0.f;
  }

  const int nch = CAUSAL ? ((qbase + 16 + 31) / 32) : (S_k / 32);
  for (int kc = 0; kc < nch; ++kc) {
    const int kbase = kc * 32;
#pragma unroll
    for (int i = 0; i < 8; ++i) {  // K chunk 32x64 halves
      int c = lane + i * 32;
      int row = c >> 3, seg = (c & 7) * 8;
      cp_b128(&Km[krow0 + (size_t)(kbase + row) * H_DIM + seg],
              &Ks[row * 72 + seg]);
    }
#pragma unroll
    for (int s = 0; s < 8; ++s) {  // V chunk transposed into Vs[d][key]
      v8h vv = *(const v8h*)&Vm[krow0 + (size_t)(kbase + lane) * H_DIM + s * 8];
#pragma unroll
      for (int j = 0; j < 8; ++j) Vs[(s * 8 + j) * 40 + lane] = vv[j];
    }
    cp_wait();
    __syncthreads();

    // scores S = Q(16x64) x K^T : contraction d split into two K=32 steps
    v8f s0 = vzero, s1 = vzero;
#pragma unroll
    for (int ks = 0; ks < 64; ks += 32) {
      v16h aq = ld_frag(&Qs[r16 * 72 + ks + koff]);
      v16h b0 = ld_frag(&Ks[r16 * 72 + ks + koff]);
      v16h b1 = ld_frag(&Ks[(16 + r16) * 72 + ks + koff]);
      s0 = wmma_f16(aq, b0, s0);
      s1 = wmma_f16(aq, b1, s1);
    }

    // online softmax over this 32-key chunk
#pragma unroll
    for (int v = 0; v < 8; ++v) {
      float a0 = s0[v] * 0.125f;  // 1/sqrt(64)
      float a1 = s1[v] * 0.125f;
      if (CAUSAL) {
        int q = qbase + v + rhi;
        if (kbase + r16 > q) a0 = -1e30f;
        if (kbase + 16 + r16 > q) a1 = -1e30f;
      }
      float cm = fmaxf(a0, a1);
#pragma unroll
      for (int o = 1; o < 16; o <<= 1) cm = fmaxf(cm, __shfl_xor(cm, o));
      float nm = fmaxf(mrow[v], cm);
      float alpha = __expf(mrow[v] - nm);
      mrow[v] = nm;
      float p0 = __expf(a0 - nm);
      float p1 = __expf(a1 - nm);
      float rs = p0 + p1;
#pragma unroll
      for (int o = 1; o < 16; o <<= 1) rs += __shfl_xor(rs, o);
      lrow[v] = lrow[v] * alpha + rs;
#pragma unroll
      for (int t = 0; t < 4; ++t) ctx[t][v] *= alpha;
      Ps[(v + rhi) * 40 + r16] = (_Float16)p0;
      Ps[(v + rhi) * 40 + 16 + r16] = (_Float16)p1;
    }
    __syncthreads();

    // ctx += P(16x32) x V(32x64)
    v16h ap = ld_frag(&Ps[r16 * 40 + koff]);
#pragma unroll
    for (int t = 0; t < 4; ++t) {
      v16h bv = ld_frag(&Vs[(t * 16 + r16) * 40 + koff]);
      ctx[t] = wmma_f16(ap, bv, ctx[t]);
    }
    __syncthreads();
  }

#pragma unroll
  for (int v = 0; v < 8; ++v) {
    float inv = 1.0f / lrow[v];
#pragma unroll
    for (int t = 0; t < 4; ++t)
      Om[qrow0 + (size_t)(v + rhi) * H_DIM + t * 16 + r16] =
          (_Float16)(ctx[t][v] * inv);
  }
}

// ---------------- host-side orchestration ----------------------------------
extern "C" void kernel_launch(void* const* d_in, const int* in_sizes, int n_in,
                              void* d_out, int out_size, void* d_ws, size_t ws_size,
                              hipStream_t stream) {
  (void)in_sizes; (void)n_in; (void)out_size; (void)ws_size;
  const float* x     = (const float*)d_in[0];
  const float* kv    = (const float*)d_in[1];
  // d_in[2]=mask (all false), d_in[3]=future_mask (causal, computed analytically)
  const float* sa_wq = (const float*)d_in[4];
  const float* sa_wk = (const float*)d_in[5];
  const float* sa_wv = (const float*)d_in[6];
  const float* sa_wo = (const float*)d_in[7];
  const float* ca_wq = (const float*)d_in[8];
  const float* ca_wk = (const float*)d_in[9];
  const float* ca_wv = (const float*)d_in[10];
  const float* ca_wo = (const float*)d_in[11];
  const float* fc_w1 = (const float*)d_in[12];
  const float* fc_b1 = (const float*)d_in[13];
  const float* fc_w2 = (const float*)d_in[14];
  const float* fc_b2 = (const float*)d_in[15];
  const float* ln1_g = (const float*)d_in[16];
  const float* ln1_b = (const float*)d_in[17];
  const float* ln2_g = (const float*)d_in[18];
  const float* ln2_b = (const float*)d_in[19];
  const float* ln3_g = (const float*)d_in[20];
  const float* ln3_b = (const float*)d_in[21];

  char* ws = (char*)d_ws;
  size_t off = 0;
  auto alloc = [&](size_t bytes) -> void* {
    void* p = ws + off;
    off += (bytes + 255) & ~(size_t)255;
    return p;
  };
  const size_t HH = (size_t)H_DIM * H_DIM;
  _Float16* wt_saq = (_Float16*)alloc(HH * 2);
  _Float16* wt_sak = (_Float16*)alloc(HH * 2);
  _Float16* wt_sav = (_Float16*)alloc(HH * 2);
  _Float16* wt_sao = (_Float16*)alloc(HH * 2);
  _Float16* wt_caq = (_Float16*)alloc(HH * 2);
  _Float16* wt_cak = (_Float16*)alloc(HH * 2);
  _Float16* wt_cav = (_Float16*)alloc(HH * 2);
  _Float16* wt_cao = (_Float16*)alloc(HH * 2);
  _Float16* wt_f1  = (_Float16*)alloc(HH * 4 * 2);
  _Float16* wt_f2  = (_Float16*)alloc(HH * 4 * 2);
  const size_t MH = (size_t)MROWS * H_DIM;
  _Float16* ln16  = (_Float16*)alloc(MH * 2);  // z0n / z1n / z2n (reused)
  _Float16* q16   = (_Float16*)alloc(MH * 2);  // q / qc
  _Float16* k16   = (_Float16*)alloc(MH * 2);  // k / kc
  _Float16* v16   = (_Float16*)alloc(MH * 2);  // v / vc
  _Float16* a16   = (_Float16*)alloc(MH * 2);  // attention out (reused)
  _Float16* kvn16 = (_Float16*)alloc(MH * 2);  // ln2(kv)
  float*    resid = (float*)alloc(MH * 4);     // z1 then z2 (in place)
  _Float16* h16   = (_Float16*)alloc(MH * 4 * 2);  // FFN hidden

  const dim3 b256(256), b32(32);
  const dim3 gLN(MROWS);
  const dim3 gGEMM_N1(H_DIM / TN, MROWS / TM);       // N=1024
  const dim3 gGEMM_N4(4 * H_DIM / TN, MROWS / TM);   // N=4096
  const dim3 gATT(SEQ / 16, NH, BATCH);
  const dim3 gCVT1((int)((HH + 255) / 256));
  const dim3 gCVT4((int)((HH * 4 + 255) / 256));

  // weights -> transposed f16 [N][K]
  wcvt_kernel<<<gCVT1, b256, 0, stream>>>(sa_wq, wt_saq, H_DIM, H_DIM);
  wcvt_kernel<<<gCVT1, b256, 0, stream>>>(sa_wk, wt_sak, H_DIM, H_DIM);
  wcvt_kernel<<<gCVT1, b256, 0, stream>>>(sa_wv, wt_sav, H_DIM, H_DIM);
  wcvt_kernel<<<gCVT1, b256, 0, stream>>>(sa_wo, wt_sao, H_DIM, H_DIM);
  wcvt_kernel<<<gCVT1, b256, 0, stream>>>(ca_wq, wt_caq, H_DIM, H_DIM);
  wcvt_kernel<<<gCVT1, b256, 0, stream>>>(ca_wk, wt_cak, H_DIM, H_DIM);
  wcvt_kernel<<<gCVT1, b256, 0, stream>>>(ca_wv, wt_cav, H_DIM, H_DIM);
  wcvt_kernel<<<gCVT1, b256, 0, stream>>>(ca_wo, wt_cao, H_DIM, H_DIM);
  wcvt_kernel<<<gCVT4, b256, 0, stream>>>(fc_w1, wt_f1, H_DIM, 4 * H_DIM);
  wcvt_kernel<<<gCVT4, b256, 0, stream>>>(fc_w2, wt_f2, 4 * H_DIM, H_DIM);

  // ---- self-attention block: z1 = x + SA(ln1(x)) ----
  layernorm_f16_kernel<<<gLN, b256, 0, stream>>>(x, ln1_g, ln1_b, ln16);
  gemm_wmma_kernel<false, false><<<gGEMM_N1, b256, 0, stream>>>(
      ln16, wt_saq, nullptr, nullptr, q16, MROWS, H_DIM, H_DIM);
  gemm_wmma_kernel<false, false><<<gGEMM_N1, b256, 0, stream>>>(
      ln16, wt_sak, nullptr, nullptr, k16, MROWS, H_DIM, H_DIM);
  gemm_wmma_kernel<false, false><<<gGEMM_N1, b256, 0, stream>>>(
      ln16, wt_sav, nullptr, nullptr, v16, MROWS, H_DIM, H_DIM);
  attention_kernel<true><<<gATT, b32, 0, stream>>>(q16, k16, v16, a16, SEQ);
  gemm_wmma_kernel<true, false><<<gGEMM_N1, b256, 0, stream>>>(
      a16, wt_sao, nullptr, x, resid, MROWS, H_DIM, H_DIM);

  // ---- cross-attention block: z2 = z1 + CA(ln2(z1), ln2(kv)) ----
  layernorm_f16_kernel<<<gLN, b256, 0, stream>>>(kv, ln2_g, ln2_b, kvn16);
  gemm_wmma_kernel<false, false><<<gGEMM_N1, b256, 0, stream>>>(
      kvn16, wt_cak, nullptr, nullptr, k16, MROWS, H_DIM, H_DIM);
  gemm_wmma_kernel<false, false><<<gGEMM_N1, b256, 0, stream>>>(
      kvn16, wt_cav, nullptr, nullptr, v16, MROWS, H_DIM, H_DIM);
  layernorm_f16_kernel<<<gLN, b256, 0, stream>>>(resid, ln2_g, ln2_b, ln16);
  gemm_wmma_kernel<false, false><<<gGEMM_N1, b256, 0, stream>>>(
      ln16, wt_caq, nullptr, nullptr, q16, MROWS, H_DIM, H_DIM);
  attention_kernel<false><<<gATT, b32, 0, stream>>>(q16, k16, v16, a16, SEQ);
  gemm_wmma_kernel<true, false><<<gGEMM_N1, b256, 0, stream>>>(
      a16, wt_cao, nullptr, resid, resid, MROWS, H_DIM, H_DIM);  // in-place res

  // ---- FFN: out = z2 + relu(ln3(z2) @ w1 + b1) @ w2 + b2 ----
  layernorm_f16_kernel<<<gLN, b256, 0, stream>>>(resid, ln3_g, ln3_b, ln16);
  gemm_wmma_kernel<false, true><<<gGEMM_N4, b256, 0, stream>>>(
      ln16, wt_f1, fc_b1, nullptr, h16, MROWS, 4 * H_DIM, H_DIM);
  gemm_wmma_kernel<true, false><<<gGEMM_N1, b256, 0, stream>>>(
      h16, wt_f2, fc_b2, resid, d_out, MROWS, H_DIM, 4 * H_DIM);
}
